// SelfAttention_Pseudoformer2_86844238725650
// MI455X (gfx1250) — matrix operand order repaired and checked by reading, not compiled, not run on hardware
//
#include <hip/hip_runtime.h>
#include <hip/hip_bf16.h>

#define B_   16
#define L_   4096
#define C_   512
#define P_   64
#define DK_  64
#define H_   8
#define HDK_ 512

typedef __attribute__((ext_vector_type(16))) __bf16 v16bf;
typedef __attribute__((ext_vector_type(8)))  float  v8f;

// round-to-nearest-even f32 -> bf16
__device__ __forceinline__ __bf16 f2bf(float f) {
    unsigned u = __builtin_bit_cast(unsigned, f);
    unsigned r = u + 0x7FFFu + ((u >> 16) & 1u);
    unsigned short h = (unsigned short)(r >> 16);
    return __builtin_bit_cast(__bf16, h);
}

// LDS byte offset of a __shared__ object (generic LDS VA low 32 bits = wave LDS offset)
__device__ __forceinline__ unsigned lds_off(const void* p) {
    return (unsigned)(unsigned long long)p;
}

// Async DMA: global (64-bit VA) -> LDS (per-lane dest offset), no VGPR round trip.
__device__ __forceinline__ void async_ld_b32(unsigned ldsdst, const float* gsrc) {
    asm volatile("global_load_async_to_lds_b32 %0, %1, off"
                 :: "v"(ldsdst), "v"(gsrc) : "memory");
}
__device__ __forceinline__ void wait_async0() {
    asm volatile("s_wait_asynccnt 0x0" ::: "memory");
}

// ---- WMMA fragment loaders -------------------------------------------------
// A: 16x32 bf16 from row-major LDS [m][k] (lda elems). element e -> K=g*16+e
__device__ __forceinline__ v16bf load_frag_a(const __bf16* As, int lda, int m0, int kk, int lane) {
    const int g  = (lane >> 4) & 1;
    const int mr = lane & 15;
    v16bf a;
#pragma unroll
    for (int e = 0; e < 16; ++e) {
        int k = (g << 4) + e;
        a[e] = As[(m0 + mr) * lda + kk + k];
    }
    return a;
}
// B: 32x16 bf16 from column-major LDS [n][k] (ldk elems). element e -> K=g*16+e (contiguous!)
__device__ __forceinline__ v16bf load_frag_bT(const __bf16* Bs, int ldk, int n0, int kk, int lane) {
    const int g  = (lane >> 4) & 1;
    const int nr = lane & 15;
    const __bf16* src = &Bs[(n0 + nr) * ldk + kk + (g << 4)];
    v16bf b;
#pragma unroll
    for (int e = 0; e < 16; ++e) b[e] = src[e];
    return b;
}
// f32-LDS variants (for async-loaded tiles), convert while building fragments
__device__ __forceinline__ v16bf load_frag_a_f32(const float* As, int lda, int m0, int kk, int lane) {
    const int g  = (lane >> 4) & 1;
    const int mr = lane & 15;
    v16bf a;
#pragma unroll
    for (int e = 0; e < 16; ++e) {
        int k = (g << 4) + e;
        a[e] = f2bf(As[(m0 + mr) * lda + kk + k]);
    }
    return a;
}
__device__ __forceinline__ v16bf load_frag_bT_f32(const float* Bs, int ldk, int n0, int kk, int lane) {
    const int g  = (lane >> 4) & 1;
    const int nr = lane & 15;
    const float* src = &Bs[(n0 + nr) * ldk + kk + (g << 4)];
    v16bf b;
#pragma unroll
    for (int e = 0; e < 16; ++e) b[e] = f2bf(src[e]);
    return b;
}

#define WMMA_BF16(a, b, c) \
    __builtin_amdgcn_wmma_f32_16x16x32_bf16(false, (a), false, (b), (short)0, (c), false, false)

// ---------------------------------------------------------------------------
// Kernel 1: pScore[b,l,p] = vx[b,l,:] . Wp[p,:] + bp[p]
// Block tile 64(l) x 64(p); 8 waves = 4m x 2n; each wave 16x32 (2 acc).
// ---------------------------------------------------------------------------
__global__ __launch_bounds__(256) void k_pscore(const float* __restrict__ vx,
                                                const float* __restrict__ Wp,
                                                const float* __restrict__ bp,
                                                float* __restrict__ pScore) {
    __shared__ __bf16 As[64 * 40];    // [m(l)][k(c)] lda=40 (padded)
    __shared__ __bf16 Bs[64 * 40];    // [n(p)][k(c)] ldk=40 (padded)
    const int b  = blockIdx.y;
    const int l0 = blockIdx.x * 64;
    const int tid = threadIdx.x, lane = tid & 31, wave = tid >> 5;
    const int m0 = (wave & 3) * 16;
    const int n0 = (wave >> 2) * 32;
    const float* vxb = vx + (size_t)b * L_ * C_;
    v8f acc0 = {}, acc1 = {};
    for (int c0 = 0; c0 < C_; c0 += 32) {
        for (int i = tid; i < 64 * 32; i += 256) {
            int k = i & 31, m = i >> 5;
            As[m * 40 + k] = f2bf(vxb[(size_t)(l0 + m) * C_ + c0 + k]);
        }
        for (int i = tid; i < 64 * 32; i += 256) {
            int k = i & 31, n = i >> 5;
            Bs[n * 40 + k] = f2bf(Wp[(size_t)n * C_ + c0 + k]);
        }
        __syncthreads();
        v16bf a  = load_frag_a(As, 40, m0, 0, lane);
        v16bf b0 = load_frag_bT(Bs, 40, n0, 0, lane);
        v16bf b1 = load_frag_bT(Bs, 40, n0 + 16, 0, lane);
        acc0 = WMMA_BF16(a, b0, acc0);
        acc1 = WMMA_BF16(a, b1, acc1);
        __syncthreads();
    }
    const int g = lane >> 4, nr = lane & 15;
    const float bv0 = bp[n0 + nr], bv1 = bp[n0 + 16 + nr];
#pragma unroll
    for (int r = 0; r < 8; ++r) {
        int m = l0 + m0 + r + 8 * g;
        pScore[((size_t)b * L_ + m) * P_ + n0 + nr]      = acc0[r] + bv0;
        pScore[((size_t)b * L_ + m) * P_ + n0 + 16 + nr] = acc1[r] + bv1;
    }
}

// ---------------------------------------------------------------------------
// Kernel 2: pAlpha[b,l,p] = softmax over l of pScore  (one block per (b,p))
// ---------------------------------------------------------------------------
__global__ __launch_bounds__(256) void k_colsoftmax(const float* __restrict__ pScore,
                                                    float* __restrict__ pAlpha) {
    const int b = blockIdx.x >> 6;
    const int p = blockIdx.x & 63;
    const int tid = threadIdx.x;
    __shared__ float red[256];
    const float* col = pScore + (size_t)b * L_ * P_ + p;
    float mx = -3.4e38f;
    for (int l = tid; l < L_; l += 256) mx = fmaxf(mx, col[(size_t)l * P_]);
    red[tid] = mx; __syncthreads();
    for (int s = 128; s > 0; s >>= 1) {
        if (tid < s) red[tid] = fmaxf(red[tid], red[tid + s]);
        __syncthreads();
    }
    mx = red[0]; __syncthreads();
    float sum = 0.f;
    for (int l = tid; l < L_; l += 256) sum += __expf(col[(size_t)l * P_] - mx);
    red[tid] = sum; __syncthreads();
    for (int s = 128; s > 0; s >>= 1) {
        if (tid < s) red[tid] += red[tid + s];
        __syncthreads();
    }
    const float inv = 1.0f / red[0];
    float* outc = pAlpha + (size_t)b * L_ * P_ + p;
    for (int l = tid; l < L_; l += 256)
        outc[(size_t)l * P_] = __expf(col[(size_t)l * P_] - mx) * inv;
}

// ---------------------------------------------------------------------------
// Kernel 3: pooled[t,b,p,c] = sum_l pAlpha[b,l,p] * X_t[b,l,c]
// Async global->LDS DMA for both tiles (fp32 staged, converted in frag build).
// Block tile 64(p) x 64(c); 8 waves = 4m x 2n; each wave 16x32 (2 acc).
// ---------------------------------------------------------------------------
__global__ __launch_bounds__(256) void k_pool(const float* __restrict__ pAlpha,
                                              const float* __restrict__ qx,
                                              const float* __restrict__ kx,
                                              const float* __restrict__ vx,
                                              float* __restrict__ pooled) {
    __shared__ float Asf[64 * 36];   // [m(p)][k(l)] lda=36 (padded)
    __shared__ float Bsf[64 * 36];   // [n(c)][k(l)] ldk=36 (padded)
    const int b  = blockIdx.z;
    const int t  = blockIdx.y;               // 0:q 1:k 2:v
    const int c0 = blockIdx.x * 64;
    const float* X = (t == 0 ? qx : (t == 1 ? kx : vx)) + (size_t)b * L_ * C_;
    const float* A = pAlpha + (size_t)b * L_ * P_;
    const int tid = threadIdx.x, lane = tid & 31, wave = tid >> 5;
    const int m0 = (wave & 3) * 16;
    const int n0 = (wave >> 2) * 32;
    v8f acc0 = {}, acc1 = {};
    for (int l0 = 0; l0 < L_; l0 += 32) {
        // DMA pAlpha tile -> Asf[p][l] (global coalesced along p, LDS scatter)
#pragma unroll
        for (int i = tid; i < 64 * 32; i += 256) {
            int p = i & 63, l = i >> 6;
            async_ld_b32(lds_off(&Asf[p * 36 + l]), &A[(size_t)(l0 + l) * P_ + p]);
        }
        // DMA X tile -> Bsf[n][k] (global coalesced along c, LDS scatter)
#pragma unroll
        for (int i = tid; i < 64 * 32; i += 256) {
            int n = i & 63, k = i >> 6;
            async_ld_b32(lds_off(&Bsf[n * 36 + k]), &X[(size_t)(l0 + k) * C_ + c0 + n]);
        }
        wait_async0();
        __syncthreads();
        v16bf a  = load_frag_a_f32(Asf, 36, m0, 0, lane);
        v16bf b0 = load_frag_bT_f32(Bsf, 36, n0, 0, lane);
        v16bf b1 = load_frag_bT_f32(Bsf, 36, n0 + 16, 0, lane);
        acc0 = WMMA_BF16(a, b0, acc0);
        acc1 = WMMA_BF16(a, b1, acc1);
        __syncthreads();
    }
    float* Y = pooled + ((size_t)t * B_ + b) * (P_ * C_);
    const int g = lane >> 4, nr = lane & 15;
#pragma unroll
    for (int r = 0; r < 8; ++r) {
        int m = m0 + r + 8 * g;
        Y[(size_t)m * C_ + c0 + n0 + nr]      = acc0[r];
        Y[(size_t)m * C_ + c0 + n0 + 16 + nr] = acc1[r];
    }
}

// ---------------------------------------------------------------------------
// Kernel 4/6: Y[b,m,n] = A[b,m,:K] . W[n,:K] + bias[n]   (M = 64 per batch)
// Block tile 64(m) x 64(n); 8 waves = 4m x 2n; each wave 16x32 (2 acc).
// ---------------------------------------------------------------------------
__global__ __launch_bounds__(256) void k_gemm64_wt(const float* __restrict__ A,
                                                   const float* __restrict__ W,
                                                   const float* __restrict__ bias,
                                                   float* __restrict__ Y,
                                                   int K, int N) {
    __shared__ __bf16 As[64 * 40];   // [m][k] lda=40
    __shared__ __bf16 Bs[64 * 40];   // [n][k] ldk=40
    const int b = blockIdx.z;
    const int n0g = blockIdx.x * 64;
    const int tid = threadIdx.x, lane = tid & 31, wave = tid >> 5;
    const int m0 = (wave & 3) * 16;
    const int n0 = (wave >> 2) * 32;
    const float* Ab = A + (size_t)b * 64 * K;
    v8f acc0 = {}, acc1 = {};
    for (int k0 = 0; k0 < K; k0 += 32) {
        for (int i = tid; i < 64 * 32; i += 256) {
            int k = i & 31, m = i >> 5;
            As[m * 40 + k] = f2bf(Ab[(size_t)m * K + k0 + k]);
        }
        for (int i = tid; i < 64 * 32; i += 256) {
            int k = i & 31, n = i >> 5;
            Bs[n * 40 + k] = f2bf(W[(size_t)(n0g + n) * K + k0 + k]);
        }
        __syncthreads();
        v16bf a  = load_frag_a(As, 40, m0, 0, lane);
        v16bf b0 = load_frag_bT(Bs, 40, n0, 0, lane);
        v16bf b1 = load_frag_bT(Bs, 40, n0 + 16, 0, lane);
        acc0 = WMMA_BF16(a, b0, acc0);
        acc1 = WMMA_BF16(a, b1, acc1);
        __syncthreads();
    }
    const int g = lane >> 4, nr = lane & 15;
    const float bv0 = bias[n0g + n0 + nr], bv1 = bias[n0g + n0 + 16 + nr];
#pragma unroll
    for (int r = 0; r < 8; ++r) {
        int m = m0 + r + 8 * g;
        Y[(size_t)b * 64 * N + (size_t)m * N + n0g + n0 + nr]      = acc0[r] + bv0;
        Y[(size_t)b * 64 * N + (size_t)m * N + n0g + n0 + 16 + nr] = acc1[r] + bv1;
    }
}

// ---------------------------------------------------------------------------
// Kernel 5: per-(b,h) 64x64 attention; writes scores output + attn result.
// ---------------------------------------------------------------------------
__global__ __launch_bounds__(64) void k_attn(const float* __restrict__ Qf,
                                             const float* __restrict__ Kf,
                                             const float* __restrict__ Vf,
                                             float* __restrict__ scoresOut,
                                             float* __restrict__ zattn) {
    const int bh = blockIdx.x;
    const int b = bh >> 3, h = bh & 7;
    __shared__ float Ks[64][64];
    __shared__ float Vs[64][64];
    __shared__ float Ss[64][65];
    const int tid = threadIdx.x;   // row p
    const float* Kb = Kf + ((size_t)b * 64) * HDK_ + h * DK_;
    const float* Vb = Vf + ((size_t)b * 64) * HDK_ + h * DK_;
#pragma unroll 8
    for (int d = 0; d < 64; ++d) {
        Ks[tid][d] = Kb[(size_t)tid * HDK_ + d];
        Vs[tid][d] = Vb[(size_t)tid * HDK_ + d];
    }
    __syncthreads();
    const float* Qrow = Qf + ((size_t)b * 64 + tid) * HDK_ + h * DK_;
    float q[64];
#pragma unroll
    for (int d = 0; d < 64; ++d) q[d] = Qrow[d];

    float mx = -3.4e38f;
    for (int j = 0; j < 64; ++j) {
        float acc = 0.f;
#pragma unroll
        for (int d = 0; d < 64; ++d) acc += q[d] * Ks[j][d];
        acc *= 0.125f;
        Ss[tid][j] = acc;
        mx = fmaxf(mx, acc);
        scoresOut[(((size_t)b * H_ + h) * 64 + tid) * 64 + j] = acc;
    }
    float sum = 0.f;
    for (int j = 0; j < 64; ++j) {
        float e = __expf(Ss[tid][j] - mx);
        Ss[tid][j] = e;
        sum += e;
    }
    const float inv = 1.0f / sum;
    float z[64];
#pragma unroll
    for (int d = 0; d < 64; ++d) z[d] = 0.f;
    for (int j = 0; j < 64; ++j) {
        float a = Ss[tid][j] * inv;
#pragma unroll
        for (int d = 0; d < 64; ++d) z[d] += a * Vs[j][d];
    }
    float* Z = zattn + ((size_t)b * 64 + tid) * HDK_ + h * DK_;
#pragma unroll
    for (int d = 0; d < 64; ++d) Z[d] = z[d];
}

// ---------------------------------------------------------------------------
// Kernel 7: z[b,l,c] = softmax_p(pScore[b,l,:]) . zmid[b,:,c]
// Block tile 32(l) x 128(c); 8 waves = 2m x 4n; each wave 16x32, K=64 (4 wmma).
// ---------------------------------------------------------------------------
__global__ __launch_bounds__(256) void k_expand(const float* __restrict__ pScore,
                                                const float* __restrict__ zmid,
                                                float* __restrict__ zout) {
    __shared__ float  Ps[32 * 65];    // padded rows
    __shared__ __bf16 As[32 * 72];    // [m(l)][k(p)] lda=72
    __shared__ __bf16 Bs[128 * 72];   // [n(c)][k(p)] ldk=72
    const int b  = blockIdx.z;
    const int c0 = blockIdx.y * 128;
    const int l0 = blockIdx.x * 32;
    const int tid = threadIdx.x, lane = tid & 31, wave = tid >> 5;
    const int m0 = (wave & 1) * 16;
    const int n0 = (wave >> 1) * 32;
    for (int i = tid; i < 32 * 64; i += 256) {
        int p = i & 63, l = i >> 6;
        Ps[l * 65 + p] = pScore[((size_t)b * L_ + l0 + l) * P_ + p];
    }
    for (int i = tid; i < 128 * 64; i += 256) {
        int n = i & 127, k = i >> 7;
        Bs[n * 72 + k] = f2bf(zmid[((size_t)b * 64 + k) * C_ + c0 + n]);
    }
    __syncthreads();
    if (tid < 32) {                        // row softmax over P=64
        float mx = -3.4e38f;
        for (int p = 0; p < 64; ++p) mx = fmaxf(mx, Ps[tid * 65 + p]);
        float sum = 0.f;
        for (int p = 0; p < 64; ++p) {
            float e = __expf(Ps[tid * 65 + p] - mx);
            Ps[tid * 65 + p] = e;
            sum += e;
        }
        const float inv = 1.0f / sum;
        for (int p = 0; p < 64; ++p) As[tid * 72 + p] = f2bf(Ps[tid * 65 + p] * inv);
    }
    __syncthreads();
    v8f acc0 = {}, acc1 = {};
#pragma unroll
    for (int kk = 0; kk < 64; kk += 32) {
        v16bf a  = load_frag_a(As, 72, m0, kk, lane);
        v16bf b0 = load_frag_bT(Bs, 72, n0, kk, lane);
        v16bf b1 = load_frag_bT(Bs, 72, n0 + 16, kk, lane);
        acc0 = WMMA_BF16(a, b0, acc0);
        acc1 = WMMA_BF16(a, b1, acc1);
    }
    const int g = lane >> 4, nr = lane & 15;
#pragma unroll
    for (int r = 0; r < 8; ++r) {
        int m = l0 + m0 + r + 8 * g;
        zout[((size_t)b * L_ + m) * C_ + c0 + n0 + nr]      = acc0[r];
        zout[((size_t)b * L_ + m) * C_ + c0 + n0 + 16 + nr] = acc1[r];
    }
}

// ---------------------------------------------------------------------------
extern "C" void kernel_launch(void* const* d_in, const int* in_sizes, int n_in,
                              void* d_out, int out_size, void* d_ws, size_t ws_size,
                              hipStream_t stream) {
    (void)in_sizes; (void)n_in; (void)out_size; (void)ws_size;
    const float* qx = (const float*)d_in[0];
    const float* kx = (const float*)d_in[1];
    const float* vx = (const float*)d_in[2];
    const float* Wp = (const float*)d_in[3];
    const float* bp = (const float*)d_in[4];
    const float* WQ = (const float*)d_in[5];
    const float* bQ = (const float*)d_in[6];
    const float* WK = (const float*)d_in[7];
    const float* bK = (const float*)d_in[8];
    const float* WV = (const float*)d_in[9];
    const float* bV = (const float*)d_in[10];
    const float* WO = (const float*)d_in[11];
    const float* bO = (const float*)d_in[12];

    // Output layout: z [B,L,C] | pScore [B,L,P] | scores [B,H,P,P]
    float* zout      = (float*)d_out;
    float* pScoreOut = zout + (size_t)B_ * L_ * C_;
    float* scoresOut = pScoreOut + (size_t)B_ * L_ * P_;

    // Workspace layout
    float* ws     = (float*)d_ws;
    float* pAlpha = ws;                                   // B*L*P
    float* pooled = pAlpha + (size_t)B_ * L_ * P_;        // 3 * B*P*C
    float* Qf     = pooled + 3 * (size_t)B_ * P_ * C_;    // B*P*HDK
    float* Kf     = Qf + (size_t)B_ * P_ * HDK_;
    float* Vf     = Kf + (size_t)B_ * P_ * HDK_;
    float* zattn  = Vf + (size_t)B_ * P_ * HDK_;
    float* zmid   = zattn + (size_t)B_ * P_ * HDK_;

    k_pscore<<<dim3(L_ / 64, B_), 256, 0, stream>>>(vx, Wp, bp, pScoreOut);
    k_colsoftmax<<<B_ * P_, 256, 0, stream>>>(pScoreOut, pAlpha);
    k_pool<<<dim3(C_ / 64, 3, B_), 256, 0, stream>>>(pAlpha, qx, kx, vx, pooled);

    const size_t bpc = (size_t)B_ * P_ * C_;
    k_gemm64_wt<<<dim3(HDK_ / 64, 1, B_), 256, 0, stream>>>(pooled + 0 * bpc, WQ, bQ, Qf, C_, HDK_);
    k_gemm64_wt<<<dim3(HDK_ / 64, 1, B_), 256, 0, stream>>>(pooled + 1 * bpc, WK, bK, Kf, C_, HDK_);
    k_gemm64_wt<<<dim3(HDK_ / 64, 1, B_), 256, 0, stream>>>(pooled + 2 * bpc, WV, bV, Vf, C_, HDK_);

    k_attn<<<B_ * H_, 64, 0, stream>>>(Qf, Kf, Vf, scoresOut, zattn);
    k_gemm64_wt<<<dim3(C_ / 64, 1, B_), 256, 0, stream>>>(zattn, WO, bO, zmid, HDK_, C_);
    k_expand<<<dim3(L_ / 32, C_ / 128, B_), 256, 0, stream>>>(pScoreOut, zmid, zout);
}
